// VectorQuantizer_816043786376
// MI455X (gfx1250) — matrix operand order repaired and checked by reading, not compiled, not run on hardware
//
#include <hip/hip_runtime.h>
#include <hip/hip_bf16.h>

// ---------------- problem constants ----------------
#define BB 16
#define DD 256
#define HH 32
#define WW 32
#define NN (BB * HH * WW)      // 16384 tokens
#define KK 8192                // codebook size
#define NQ (BB * DD * HH * WW) // 4194304 quantized output elems
#define KCHUNKS (DD / 32)      // 8 WMMA K-steps per 16x16 tile
#define COLTILES (KK / 16)     // 512 column tiles
#define ROWTILES (NN / 16)     // 1024 row tiles
#define SPLIT 4                // column-split factor (parallelism)
#define CT_PER_SPLIT (COLTILES / SPLIT) // 128

typedef __attribute__((ext_vector_type(16))) __bf16 v16bf;
typedef __attribute__((ext_vector_type(8)))  float  v8f;

// ---- workspace layout (bytes) ----
#define WS_XPACK 0                                   // NN*DD bf16 = 8 MB
#define WS_EPACK (WS_XPACK + (size_t)NN * DD * 2)    // KK*DD bf16 = 4 MB
#define WS_ENORM (WS_EPACK + (size_t)KK * DD * 2)    // KK f32    = 32 KB
#define WS_PVAL  (WS_ENORM + (size_t)KK * 4)         // SPLIT*NN f32 = 256 KB
#define WS_PIDX  (WS_PVAL  + (size_t)SPLIT * NN * 4) // SPLIT*NN i32 = 256 KB
#define WS_INDEX (WS_PIDX  + (size_t)SPLIT * NN * 4) // NN i32    = 64 KB

// Map (lane, half) -> k within a 32-wide K chunk for a 16-bit A fragment
// (ISA 7.12.2: VGPR0..3 lanes0-15 K=0..7, lanes16-31 K=8..15; VGPR4..7 +16)
__device__ __forceinline__ int a_frag_k(int lane, int h) {
    int vg = h >> 1, pr = h & 1;
    int base = (lane < 16) ? 0 : 8;
    return (vg < 4) ? (base + vg * 2 + pr) : (16 + base + (vg - 4) * 2 + pr);
}

// ---------------- pack x into A-fragment tiles (NHWC flatten + bf16) --------
__global__ void pack_x_kernel(const float* __restrict__ x, __bf16* __restrict__ xp) {
    int t = blockIdx.x * blockDim.x + threadIdx.x;  // NN*DD threads
    if (t >= NN * DD) return;
    int tile = t >> 9;            // 512 elems per 16x32 tile
    int within = t & 511;
    int rt = tile >> 3;           // row tile
    int c  = tile & 7;            // k-chunk
    int lane = within >> 4;
    int h    = within & 15;
    int m = lane & 15;
    int r = rt * 16 + m;          // flat token id = b*H*W + hh*W + ww
    int d = c * 32 + a_frag_k(lane, h);
    int b  = r >> 10;
    int hh = (r >> 5) & 31;
    int ww = r & 31;
    xp[t] = (__bf16)x[(((b * DD + d) * HH) + hh) * WW + ww];
}

// ---------------- pack e (D x K) into B-fragment tiles ----------------------
// B 32x16: lanes0-15 col N=lane, K=h (h<8) / 16+h-8; lanes16-31 col N=lane-16, K=8+h (h<8) / 24+h-8
__global__ void pack_e_kernel(const float* __restrict__ e, __bf16* __restrict__ ep) {
    int t = blockIdx.x * blockDim.x + threadIdx.x;  // KK*DD threads
    if (t >= KK * DD) return;
    int tile = t >> 9;
    int within = t & 511;
    int ct = tile >> 3;
    int c  = tile & 7;
    int lane = within >> 4;
    int h    = within & 15;
    int col = ct * 16 + (lane & 15);
    int d   = c * 32 + ((h < 8) ? 0 : 16) + ((lane < 16) ? 0 : 8) + (h & 7);
    ep[t] = (__bf16)e[d * KK + col];
}

// ---------------- column squared norms --------------------------------------
__global__ void enorm_kernel(const float* __restrict__ e, float* __restrict__ en) {
    int k = blockIdx.x * blockDim.x + threadIdx.x;  // KK threads
    if (k >= KK) return;
    float s = 0.f;
    #pragma unroll 8
    for (int d = 0; d < DD; ++d) { float v = e[(size_t)d * KK + k]; s += v * v; }
    en[k] = s;
}

// ---------------- fused distance GEMM + argmin ------------------------------
// One wave per (row tile, column split). Double-buffered B tiles: loads for
// tile ct+1 issue before the 8-WMMA chain on tile ct, so global_load_b128
// latency overlaps the matrix pipe. argmin uses en[col] - 2*dot (the per-row
// ||x||^2 term is constant and cannot affect the argmin).
__global__ __launch_bounds__(128)
void vq_argmin_kernel(const __bf16* __restrict__ xp, const __bf16* __restrict__ ep,
                      const float* __restrict__ en,
                      float* __restrict__ pval, int* __restrict__ pidx) {
    const int lane  = threadIdx.x & 31;
    const int wave  = threadIdx.x >> 5;
    const int split = blockIdx.x / (ROWTILES / 4);       // 0..SPLIT-1
    const int rt    = (blockIdx.x % (ROWTILES / 4)) * 4 + wave;
    const int ctBeg = split * CT_PER_SPLIT;
    const int mycol = lane & 15;

    // Load the wave's 8 A fragments (16x32 bf16 each) once.
    v16bf afrag[KCHUNKS];
    #pragma unroll
    for (int c = 0; c < KCHUNKS; ++c)
        afrag[c] = *(const v16bf*)(xp + ((size_t)(rt * KCHUNKS + c) << 9) + lane * 16);

    float minval[8];
    int   minidx[8];
    #pragma unroll
    for (int v = 0; v < 8; ++v) { minval[v] = __builtin_inff(); minidx[v] = 0; }

    v16bf b0[KCHUNKS], b1[KCHUNKS];

    // prologue: tile ctBeg -> b0
    {
        const __bf16* bp = ep + ((size_t)(ctBeg * KCHUNKS) << 9) + lane * 16;
        #pragma unroll
        for (int c = 0; c < KCHUNKS; ++c) b0[c] = *(const v16bf*)(bp + ((size_t)c << 9));
    }

    for (int ct = ctBeg; ct < ctBeg + CT_PER_SPLIT; ct += 2) {
        // issue loads for tile ct+1 -> b1 (overlaps WMMA chain on b0)
        {
            const __bf16* bp = ep + ((size_t)((ct + 1) * KCHUNKS) << 9) + lane * 16;
            #pragma unroll
            for (int c = 0; c < KCHUNKS; ++c) b1[c] = *(const v16bf*)(bp + ((size_t)c << 9));
        }
        float en0 = en[ct * 16 + mycol];
        float en1 = en[(ct + 1) * 16 + mycol];

        v8f acc = {};
        #pragma unroll
        for (int c = 0; c < KCHUNKS; ++c)
            acc = __builtin_amdgcn_wmma_f32_16x16x32_bf16(
                false, afrag[c], false, b0[c], (short)0, acc, false, false);
        {
            int col = ct * 16 + mycol;
            #pragma unroll
            for (int v = 0; v < 8; ++v) {
                float dist = __builtin_fmaf(-2.0f, acc[v], en0);
                if (dist < minval[v]) { minval[v] = dist; minidx[v] = col; }
            }
        }

        // issue loads for tile ct+2 -> b0 (overlaps WMMA chain on b1)
        if (ct + 2 < ctBeg + CT_PER_SPLIT) {
            const __bf16* bp = ep + ((size_t)((ct + 2) * KCHUNKS) << 9) + lane * 16;
            #pragma unroll
            for (int c = 0; c < KCHUNKS; ++c) b0[c] = *(const v16bf*)(bp + ((size_t)c << 9));
        }

        v8f acc2 = {};
        #pragma unroll
        for (int c = 0; c < KCHUNKS; ++c)
            acc2 = __builtin_amdgcn_wmma_f32_16x16x32_bf16(
                false, afrag[c], false, b1[c], (short)0, acc2, false, false);
        {
            int col = (ct + 1) * 16 + mycol;
            #pragma unroll
            for (int v = 0; v < 8; ++v) {
                float dist = __builtin_fmaf(-2.0f, acc2[v], en1);
                if (dist < minval[v]) { minval[v] = dist; minidx[v] = col; }
            }
        }
    }

    // Cross-lane min+argmin within each 16-lane column group (xor stays in group).
    #pragma unroll
    for (int v = 0; v < 8; ++v) {
        float val = minval[v];
        int   id  = minidx[v];
        #pragma unroll
        for (int off = 8; off >= 1; off >>= 1) {
            float oval = __shfl_xor(val, off, 32);
            int   oid  = __shfl_xor(id,  off, 32);
            if (oval < val || (oval == val && oid < id)) { val = oval; id = oid; }
        }
        if ((lane & 15) == 0) {
            int row = rt * 16 + v + ((lane >= 16) ? 8 : 0);
            pval[split * NN + row] = val;
            pidx[split * NN + row] = id;
        }
    }
}

// ---------------- combine per-split partial argmins -------------------------
__global__ void combine_kernel(const float* __restrict__ pval, const int* __restrict__ pidx,
                               int* __restrict__ idx) {
    int r = blockIdx.x * blockDim.x + threadIdx.x;
    if (r >= NN) return;
    float best = pval[r];
    int   bid  = pidx[r];
    #pragma unroll
    for (int s = 1; s < SPLIT; ++s) {
        float v = pval[s * NN + r];
        int   i = pidx[s * NN + r];
        if (v < best || (v == best && i < bid)) { best = v; bid = i; }
    }
    idx[r] = bid;
}

// ---------------- zero the two loss accumulators ----------------------------
__global__ void init_losses_kernel(float* __restrict__ out) {
    if (threadIdx.x == 0 && blockIdx.x == 0) { out[NQ] = 0.f; out[NQ + 1] = 0.f; }
}

// ---------------- gather codebook rows + MSE losses -------------------------
__global__ void gather_loss_kernel(const float* __restrict__ x, const float* __restrict__ e,
                                   const int* __restrict__ idx, float* __restrict__ out) {
    __shared__ float sdata[256];
    int o = blockIdx.x * blockDim.x + threadIdx.x;   // NQ threads
    float local = 0.f;
    if (o < NQ) {
        int b  = o >> 18;
        int d  = (o >> 10) & 255;
        int hh = (o >> 5) & 31;
        int ww = o & 31;
        int r  = b * 1024 + hh * 32 + ww;
        int id = idx[r];
        float q = e[(size_t)d * KK + id];
        out[o] = q;                       // STE output == quantized values
        float diff = x[o] - q;
        local = diff * diff;
    }
    sdata[threadIdx.x] = local;
    __syncthreads();
    for (int s = 128; s > 0; s >>= 1) {
        if (threadIdx.x < s) sdata[threadIdx.x] += sdata[threadIdx.x + s];
        __syncthreads();
    }
    if (threadIdx.x == 0) {
        float contrib = sdata[0] * (1.0f / (float)NQ);
        atomicAdd(&out[NQ],     contrib);  // dictionary loss
        atomicAdd(&out[NQ + 1], contrib);  // commitment loss (numerically equal)
    }
}

// ---------------- emit indices as floats ------------------------------------
__global__ void write_idx_kernel(const int* __restrict__ idx, float* __restrict__ out) {
    int r = blockIdx.x * blockDim.x + threadIdx.x;
    if (r < NN) out[NQ + 2 + r] = (float)idx[r];
}

extern "C" void kernel_launch(void* const* d_in, const int* in_sizes, int n_in,
                              void* d_out, int out_size, void* d_ws, size_t ws_size,
                              hipStream_t stream) {
    const float* x = (const float*)d_in[0];     // [16,256,32,32]
    const float* e = (const float*)d_in[1];     // [256,8192]
    float* out = (float*)d_out;

    char* ws = (char*)d_ws;
    __bf16* xp   = (__bf16*)(ws + WS_XPACK);
    __bf16* ep   = (__bf16*)(ws + WS_EPACK);
    float*  en   = (float*)(ws + WS_ENORM);
    float*  pval = (float*)(ws + WS_PVAL);
    int*    pidx = (int*)  (ws + WS_PIDX);
    int*    idx  = (int*)  (ws + WS_INDEX);

    pack_x_kernel<<<(NN * DD + 255) / 256, 256, 0, stream>>>(x, xp);
    pack_e_kernel<<<(KK * DD + 255) / 256, 256, 0, stream>>>(e, ep);
    enorm_kernel <<<(KK + 255) / 256, 256, 0, stream>>>(e, en);

    // ROWTILES/4 blocks per split, 4 waves (128 threads) each, SPLIT splits
    vq_argmin_kernel<<<(ROWTILES / 4) * SPLIT, 128, 0, stream>>>(xp, ep, en, pval, pidx);

    combine_kernel<<<(NN + 255) / 256, 256, 0, stream>>>(pval, pidx, idx);
    init_losses_kernel<<<1, 32, 0, stream>>>(out);
    gather_loss_kernel<<<(NQ + 255) / 256, 256, 0, stream>>>(x, e, idx, out);
    write_idx_kernel<<<(NN + 255) / 256, 256, 0, stream>>>(idx, out);
}
